// BigramLanguageModel_85727547228746
// MI455X (gfx1250) — matrix operand order repaired and checked by reading, not compiled, not run on hardware
//
#include <hip/hip_runtime.h>
#include <hip/hip_bf16.h>

// ============================================================================
// Transformer (enc-dec) forward for MI455X / gfx1250, wave32 + WMMA bf16.
//
// All GEMMs run on v_wmma_f32_16x16x32_bf16 (fp32 accumulate). f32 inputs are
// converted to bf16 on the fly (LDS-staged, double-buffered for the big
// GEMMs). LN / softmax / residual / bias epilogues stay fp32.
//
// Input pointer order assumes JAX pytree flattening (sorted dict keys at every
// level) of setup_inputs():
//   0  ca_mask                     [B,T,T] i32
//   1..26  dec_params: ca.{Wk,Wo,Wq,Wv,bo}, ffn.{W1,W2,b1,b2}, lmW, lmb,
//          ln1.{b,s}, ln2.{b,s}, ln3.{b,s}, lnf_b, lnf_s, pos,
//          sa.{Wk,Wo,Wq,Wv,bo}, tok
//   27..41 enc_params: ffn.{W1,W2,b1,b2}, ln1.{b,s}, ln2.{b,s}, pos,
//          sa.{Wk,Wo,Wq,Wv,bo}, tok
//   42 x, 43 x_mask, 44 y, 45 y_mask
// ============================================================================

typedef __bf16 bf16;
typedef __attribute__((ext_vector_type(16))) __bf16 bf16x16;
typedef __attribute__((ext_vector_type(8)))  float  f32x8;

constexpr int V  = 100280;
constexpr int E  = 512;
constexpr int T  = 256;
constexpr int B  = 4;
constexpr int H  = 8;
constexpr int HD = 64;
constexpr int L  = 10;
constexpr int FF = 2048;
constexpr int BT = B * T;   // 1024
constexpr int BH = B * H;   // 32
constexpr float SCALE = 0.04419417382415922f;  // 512^-0.5 (faithful to reference)

// ---------------------------------------------------------------------------
// WMMA fragment loaders (wave32; contiguous k order used for both A and B, not the ISA table order).
// A 16x32 bf16: lane L -> M=L&15, holds K {k16..k16+15},
// k16 = (L>=16)?16:0.  B 32x16 bf16: lane L -> N=L&15, holds K
// {k16..k16+15}, k16 = (L>=16)?16:0 (contiguous when B stored [N][K]).
// ---------------------------------------------------------------------------
__device__ __forceinline__ bf16x16 load_frag_a(const bf16* __restrict__ p) {
  bf16x16 v;
#pragma unroll
  for (int i = 0; i < 8; ++i) v[i] = p[i];
#pragma unroll
  for (int i = 0; i < 8; ++i) v[8 + i] = p[8 + i];
  return v;
}
__device__ __forceinline__ bf16x16 load_frag_b(const bf16* __restrict__ p) {
  bf16x16 v;
#pragma unroll
  for (int i = 0; i < 16; ++i) v[i] = p[i];
  return v;
}

// ---------------------------------------------------------------------------
// Embedding: out[b,t,:] = tok[idx[b,t],:] + pos[t,:]
// ---------------------------------------------------------------------------
__global__ __launch_bounds__(256)
void embed_kernel(const int* __restrict__ idx, const float* __restrict__ tok,
                  const float* __restrict__ pos, float* __restrict__ out) {
  size_t i = (size_t)blockIdx.x * 256 + threadIdx.x;  // over BT*E
  int e = (int)(i % E);
  size_t bt = i / E;
  int t = (int)(bt % T);
  out[i] = tok[(size_t)idx[bt] * E + e] + pos[(size_t)t * E + e];
}

// ---------------------------------------------------------------------------
// LayerNorm over E=512, one block (256 thr) per row, fp32.
// ---------------------------------------------------------------------------
__global__ __launch_bounds__(256)
void layernorm_kernel(const float* __restrict__ X, const float* __restrict__ g,
                      const float* __restrict__ be, float* __restrict__ Y) {
  const int row = blockIdx.x, t = threadIdx.x;
  const float* x = X + (size_t)row * E;
  float a = x[t], c = x[t + 256];
  __shared__ float red[256];
  red[t] = a + c;
  __syncthreads();
#pragma unroll
  for (int w2 = 128; w2 > 0; w2 >>= 1) {
    if (t < w2) red[t] += red[t + w2];
    __syncthreads();
  }
  const float mean = red[0] * (1.0f / E);
  __syncthreads();
  const float da = a - mean, dc = c - mean;
  red[t] = da * da + dc * dc;
  __syncthreads();
#pragma unroll
  for (int w2 = 128; w2 > 0; w2 >>= 1) {
    if (t < w2) red[t] += red[t + w2];
    __syncthreads();
  }
  const float inv = rsqrtf(red[0] * (1.0f / E) + 1e-5f);
  float* y = Y + (size_t)row * E;
  y[t]       = g[t] * da * inv + be[t];
  y[t + 256] = g[t + 256] * dc * inv + be[t + 256];
}

// ---------------------------------------------------------------------------
// Generic GEMM: out[M,N] = act( A[M,K] @ W[K,N] + bias ) (+ res).
// Block tile 128x128, BK=32, double-buffered LDS, 256 threads = 8 waves
// (4x2), each wave 32x64 (2x4 WMMA tiles -> 8 v_wmma per K-step).
// EDGE template: N-tail guards only in the (single) remainder column launch,
// so the hot path has no exec-mask branching.
// ---------------------------------------------------------------------------
constexpr int BMt = 128, BNt = 128, BKt = 32, LDP = BKt + 8;  // LDS row pad

template <bool EDGE>
__global__ __launch_bounds__(256)
void gemm_bf16_kernel(const float* __restrict__ A, const float* __restrict__ W,
                      const float* __restrict__ bias, const float* __restrict__ res,
                      float* __restrict__ out, int M, int K, int N, int relu,
                      int n_block_off) {
  __shared__ bf16 sA[2][BMt][LDP];
  __shared__ bf16 sB[2][BNt][LDP];  // W tile stored transposed: sB[..][n][k]

  const int tid  = threadIdx.x;
  const int lane = tid & 31;
  const int wid  = tid >> 5;
  const int m0   = blockIdx.y * BMt;
  const int n0   = (blockIdx.x + n_block_off) * BNt;
  const int wm   = (wid & 3) * 32;
  const int wn   = (wid >> 2) * 64;

  // per-thread staging coordinates
  const int ar = tid >> 3, ac = (tid & 7) * 4;     // A: 128x32, 4 rounds of 32 rows
  const int wr = tid >> 5, wc = (tid & 31) * 4;    // W: 32x128, 4 rounds of 8 rows

  f32x8 acc[2][4] = {};
  float4 ra[4], rb[4];

  auto load_regs = [&](int k0) {
#pragma unroll
    for (int it = 0; it < 4; ++it)
      ra[it] = *(const float4*)(A + (size_t)(m0 + ar + it * 32) * K + (k0 + ac));
#pragma unroll
    for (int it = 0; it < 4; ++it) {
      const int gr = k0 + wr + it * 8;
      const int gn = n0 + wc;
      if (!EDGE || gn + 3 < N) {
        rb[it] = *(const float4*)(W + (size_t)gr * N + gn);
      } else {
        rb[it].x = (gn + 0 < N) ? W[(size_t)gr * N + gn + 0] : 0.0f;
        rb[it].y = (gn + 1 < N) ? W[(size_t)gr * N + gn + 1] : 0.0f;
        rb[it].z = (gn + 2 < N) ? W[(size_t)gr * N + gn + 2] : 0.0f;
        rb[it].w = (gn + 3 < N) ? W[(size_t)gr * N + gn + 3] : 0.0f;
      }
    }
  };
  auto store_lds = [&](int buf) {
#pragma unroll
    for (int it = 0; it < 4; ++it) {
      const int r = ar + it * 32;
      sA[buf][r][ac + 0] = (bf16)ra[it].x; sA[buf][r][ac + 1] = (bf16)ra[it].y;
      sA[buf][r][ac + 2] = (bf16)ra[it].z; sA[buf][r][ac + 3] = (bf16)ra[it].w;
      const int r2 = wr + it * 8;
      sB[buf][wc + 0][r2] = (bf16)rb[it].x; sB[buf][wc + 1][r2] = (bf16)rb[it].y;
      sB[buf][wc + 2][r2] = (bf16)rb[it].z; sB[buf][wc + 3][r2] = (bf16)rb[it].w;
    }
  };

  const int nk = K / BKt;
  load_regs(0);
  store_lds(0);
  __syncthreads();

  for (int kt = 0; kt < nk; ++kt) {
    const int cur = kt & 1;
    if (kt + 1 < nk) {
      load_regs((kt + 1) * BKt);  // global loads overlap the WMMAs below
      __builtin_prefetch(A + (size_t)(m0 + ar) * K + (kt + 2) * BKt, 0, 1);
    }

    bf16x16 af[2], bfv[4];
#pragma unroll
    for (int tm = 0; tm < 2; ++tm)
      af[tm] = load_frag_a(&sA[cur][wm + tm * 16 + (lane & 15)][(lane >> 4) * 16]);
#pragma unroll
    for (int tn = 0; tn < 4; ++tn)
      bfv[tn] = load_frag_b(&sB[cur][wn + tn * 16 + (lane & 15)][(lane >> 4) * 16]);
#pragma unroll
    for (int tm = 0; tm < 2; ++tm)
#pragma unroll
      for (int tn = 0; tn < 4; ++tn)
        acc[tm][tn] = __builtin_amdgcn_wmma_f32_16x16x32_bf16(
            false, af[tm], false, bfv[tn], (short)0, acc[tm][tn], false, false);

    if (kt + 1 < nk) store_lds(cur ^ 1);
    __syncthreads();
  }

  // ---- epilogue: C layout VGPR i -> M = base + i, N = lane&15 ----
#pragma unroll
  for (int tm = 0; tm < 2; ++tm) {
#pragma unroll
    for (int tn = 0; tn < 4; ++tn) {
      const int n = n0 + wn + tn * 16 + (lane & 15);
      if (!EDGE || n < N) {
        const int mbase = m0 + wm + tm * 16 + (lane >> 4) * 8;
        const float bv = bias ? bias[n] : 0.0f;
#pragma unroll
        for (int i = 0; i < 8; ++i) {
          float vv = acc[tm][tn][i] + bv;
          if (relu) vv = fmaxf(vv, 0.0f);
          if (res) vv += res[(size_t)(mbase + i) * N + n];
          out[(size_t)(mbase + i) * N + n] = vv;
        }
      }
    }
  }
}

// ---------------------------------------------------------------------------
// Head permute (fused q,k,v): f32 [B,T,E] -> bf16 [B*H,T,HD] (q,k) and
// [B*H,HD,T] (v^T).
// ---------------------------------------------------------------------------
__global__ __launch_bounds__(256)
void heads_permute_kernel(const float* __restrict__ Q, const float* __restrict__ K,
                          const float* __restrict__ Vv, bf16* __restrict__ Qb,
                          bf16* __restrict__ Kb, bf16* __restrict__ Vtb) {
  size_t i = (size_t)blockIdx.x * 256 + threadIdx.x;
  const int e = (int)(i % E);
  const size_t bt = i / E;
  const int t = (int)(bt % T), b = (int)(bt / T);
  const int h = e / HD, d = e % HD;
  const size_t qk = (((size_t)(b * H + h)) * T + t) * HD + d;
  Qb[qk] = (bf16)Q[i];
  Kb[qk] = (bf16)K[i];
  Vtb[(((size_t)(b * H + h)) * HD + d) * T + t] = (bf16)Vv[i];
}

// ---------------------------------------------------------------------------
// S = scale * Q K^T : 8 waves/block, each wave one 16x16 tile, K=HD=64.
// Q,K bf16 [BH,T,HD]; S f32 [BH,T,T].  Grid: (T/16*T/16/8, BH).
// ---------------------------------------------------------------------------
__global__ __launch_bounds__(256)
void attn_score_kernel(const bf16* __restrict__ Q, const bf16* __restrict__ Kb,
                       float* __restrict__ S) {
  const int lane = threadIdx.x & 31;
  const int tile = blockIdx.x * 8 + (threadIdx.x >> 5);  // 0..255
  const int mt = tile >> 4, nt = tile & 15, bh = blockIdx.y;
  const bf16* qp = Q  + ((size_t)bh * T + mt * 16 + (lane & 15)) * HD + (lane >> 4) * 16;
  const bf16* kp = Kb + ((size_t)bh * T + nt * 16 + (lane & 15)) * HD + (lane >> 4) * 16;
  f32x8 acc = {};
#pragma unroll
  for (int ks = 0; ks < HD; ks += 32) {
    bf16x16 a = load_frag_a(qp + ks);
    bf16x16 b = load_frag_b(kp + ks);
    acc = __builtin_amdgcn_wmma_f32_16x16x32_bf16(false, a, false, b, (short)0,
                                                  acc, false, false);
  }
  float* sp = S + ((size_t)bh * T + mt * 16) * T + nt * 16;
  const int n = lane & 15;
  const int mb = (lane >> 4) * 8;
#pragma unroll
  for (int i = 0; i < 8; ++i) sp[(size_t)(mb + i) * T + n] = acc[i] * SCALE;
}

// ---------------------------------------------------------------------------
// Masked softmax over T=256, one block per (bh,row). mask [B,T,T] i32.
// causal adds n<=m constraint (decoder self-attn). Probs written bf16.
// ---------------------------------------------------------------------------
__global__ __launch_bounds__(256)
void softmax_kernel(const float* __restrict__ S, const int* __restrict__ mask,
                    int causal, bf16* __restrict__ P) {
  const int m = blockIdx.x, bh = blockIdx.y, b = bh / H, n = threadIdx.x;
  const size_t srow = ((size_t)bh * T + m) * T;
  float s = S[srow + n];
  int mk = mask[((size_t)b * T + m) * T + n];
  if (causal && n > m) mk = 0;
  s = mk ? s : -1e9f;
  __shared__ float red[T];
  red[n] = s;
  __syncthreads();
#pragma unroll
  for (int w2 = T / 2; w2 > 0; w2 >>= 1) {
    if (n < w2) red[n] = fmaxf(red[n], red[n + w2]);
    __syncthreads();
  }
  const float mx = red[0];
  __syncthreads();
  const float e = __expf(s - mx);
  red[n] = e;
  __syncthreads();
#pragma unroll
  for (int w2 = T / 2; w2 > 0; w2 >>= 1) {
    if (n < w2) red[n] += red[n + w2];
    __syncthreads();
  }
  P[srow + n] = (bf16)(e / red[0]);
}

// ---------------------------------------------------------------------------
// O = P V : 8 waves/block, each wave one 16x16 tile, K=T=256 (8 WMMA steps).
// P bf16 [BH,T,T], V^T bf16 [BH,HD,T]; O f32 [B,T,E].  Grid: (8, BH).
// ---------------------------------------------------------------------------
__global__ __launch_bounds__(256)
void attn_out_kernel(const bf16* __restrict__ P, const bf16* __restrict__ Vt,
                     float* __restrict__ O) {
  const int lane = threadIdx.x & 31;
  const int tile = blockIdx.x * 8 + (threadIdx.x >> 5);  // 0..63
  const int mt = tile >> 2, nt = tile & 3, bh = blockIdx.y;
  const bf16* pp = P  + ((size_t)bh * T + mt * 16 + (lane & 15)) * T + (lane >> 4) * 16;
  const bf16* vp = Vt + ((size_t)bh * HD + nt * 16 + (lane & 15)) * T + (lane >> 4) * 16;
  f32x8 acc = {};
#pragma unroll
  for (int ks = 0; ks < T; ks += 32) {
    bf16x16 a = load_frag_a(pp + ks);
    bf16x16 b = load_frag_b(vp + ks);
    acc = __builtin_amdgcn_wmma_f32_16x16x32_bf16(false, a, false, b, (short)0,
                                                  acc, false, false);
  }
  const int bb = bh / H, hh = bh % H;
  const int d = nt * 16 + (lane & 15);
  const int mb = mt * 16 + (lane >> 4) * 8;
#pragma unroll
  for (int i = 0; i < 8; ++i)
    O[((size_t)bb * T + mb + i) * E + hh * HD + d] = acc[i];
}

// ===========================================================================
// Host orchestration
// ===========================================================================
extern "C" void kernel_launch(void* const* d_in, const int* in_sizes, int n_in,
                              void* d_out, int out_size, void* d_ws, size_t ws_size,
                              hipStream_t stream) {
  (void)in_sizes; (void)n_in; (void)out_size; (void)ws_size;
  auto F = [&](int i) { return (const float*)d_in[i]; };
  auto I = [&](int i) { return (const int*)d_in[i]; };

  // -------- workspace carve-out (~39 MB) --------
  char* wp = (char*)d_ws;
  auto alloc = [&](size_t bytes) -> void* {
    void* p = wp;
    wp += (bytes + 255) & ~(size_t)255;
    return p;
  };
  float* hbuf   = (float*)alloc((size_t)BT * E * 4);
  float* encout = (float*)alloc((size_t)BT * E * 4);
  float* lnbuf  = (float*)alloc((size_t)BT * E * 4);
  float* qf     = (float*)alloc((size_t)BT * E * 4);
  float* kf     = (float*)alloc((size_t)BT * E * 4);
  float* vf     = (float*)alloc((size_t)BT * E * 4);
  float* attno  = (float*)alloc((size_t)BT * E * 4);
  bf16*  qb     = (bf16*) alloc((size_t)BT * E * 2);
  bf16*  kb     = (bf16*) alloc((size_t)BT * E * 2);
  bf16*  vtb    = (bf16*) alloc((size_t)BT * E * 2);
  float* scores = (float*)alloc((size_t)BH * T * T * 4);
  bf16*  probs  = (bf16*) alloc((size_t)BH * T * T * 2);
  float* ffmid  = (float*)alloc((size_t)BT * FF * 4);

  auto gemm = [&](const float* A, const float* W, const float* bias,
                  const float* res, float* out, int M, int K, int N, int relu) {
    const int full = N / BNt;
    if (full > 0)
      gemm_bf16_kernel<false><<<dim3(full, M / BMt), 256, 0, stream>>>(
          A, W, bias, res, out, M, K, N, relu, 0);
    if (N % BNt)
      gemm_bf16_kernel<true><<<dim3(1, M / BMt), 256, 0, stream>>>(
          A, W, bias, res, out, M, K, N, relu, full);
  };
  auto lnorm = [&](const float* X, const float* s, const float* b, float* Y) {
    layernorm_kernel<<<BT, 256, 0, stream>>>(X, s, b, Y);
  };
  auto mha = [&](const float* q_in, const float* kv_in, const int* mask,
                 int causal, const float* Wq, const float* Wk, const float* Wv,
                 const float* Wo, const float* bo, float* hio) {
    gemm(q_in,  Wq, nullptr, nullptr, qf, BT, E, E, 0);
    gemm(kv_in, Wk, nullptr, nullptr, kf, BT, E, E, 0);
    gemm(kv_in, Wv, nullptr, nullptr, vf, BT, E, E, 0);
    heads_permute_kernel<<<dim3(BT * E / 256), 256, 0, stream>>>(qf, kf, vf,
                                                                 qb, kb, vtb);
    attn_score_kernel<<<dim3(T / 16 * T / 16 / 8, BH), 256, 0, stream>>>(qb, kb, scores);
    softmax_kernel<<<dim3(T, BH), T, 0, stream>>>(scores, mask, causal, probs);
    attn_out_kernel<<<dim3(T / 16 * HD / 16 / 8, BH), 256, 0, stream>>>(probs, vtb, attno);
    gemm(attno, Wo, bo, hio, hio, BT, E, E, 0);  // hio += attno @ Wo + bo
  };
  auto ffn = [&](const float* W1, const float* b1, const float* W2,
                 const float* b2, float* hio) {
    gemm(lnbuf, W1, b1, nullptr, ffmid, BT, E, FF, 1);  // relu(x@W1+b1)
    gemm(ffmid, W2, b2, hio, hio, BT, FF, E, 0);        // hio += .@W2+b2
  };

  // ---------------- encoder (input y) ----------------
  embed_kernel<<<dim3(BT * E / 256), 256, 0, stream>>>(I(44), F(41), F(35), hbuf);
  for (int l = 0; l < L; ++l) {
    const size_t EEo = (size_t)l * E * E, Eo = (size_t)l * E,
                 EFo = (size_t)l * E * FF, Fo = (size_t)l * FF;
    lnorm(hbuf, F(32) + Eo, F(31) + Eo, lnbuf);                    // enc.ln1
    mha(lnbuf, lnbuf, I(45), 0,                                    // y_mask
        F(38) + EEo, F(36) + EEo, F(39) + EEo, F(37) + EEo, F(40) + Eo, hbuf);
    lnorm(hbuf, F(34) + Eo, F(33) + Eo, lnbuf);                    // enc.ln2
    ffn(F(27) + EFo, F(29) + Fo, F(28) + EFo, F(30) + Eo, hbuf);
  }
  hipMemcpyAsync(encout, hbuf, (size_t)BT * E * 4, hipMemcpyDeviceToDevice, stream);

  // ---------------- decoder (input x) ----------------
  embed_kernel<<<dim3(BT * E / 256), 256, 0, stream>>>(I(42), F(26), F(20), hbuf);
  for (int l = 0; l < L; ++l) {
    const size_t EEo = (size_t)l * E * E, Eo = (size_t)l * E,
                 EFo = (size_t)l * E * FF, Fo = (size_t)l * FF;
    lnorm(hbuf, F(13) + Eo, F(12) + Eo, lnbuf);                    // dec.ln1
    mha(lnbuf, lnbuf, I(43), 1,                                    // x_mask & tril
        F(23) + EEo, F(21) + EEo, F(24) + EEo, F(22) + EEo, F(25) + Eo, hbuf);
    lnorm(hbuf, F(15) + Eo, F(14) + Eo, lnbuf);                    // dec.ln2
    mha(lnbuf, encout, I(0), 0,                                    // ca_mask
        F(3) + EEo, F(1) + EEo, F(4) + EEo, F(2) + EEo, F(5) + Eo, hbuf);
    lnorm(hbuf, F(17) + Eo, F(16) + Eo, lnbuf);                    // dec.ln3
    ffn(F(6) + EFo, F(8) + Fo, F(7) + EFo, F(9) + Eo, hbuf);
  }

  // ---------------- LM head ----------------
  lnorm(hbuf, F(19), F(18), lnbuf);                                // lnf
  gemm(lnbuf, F(10), F(11), nullptr, (float*)d_out, BT, E, V, 0);  // lmW, lmb
}
